// MultiHeadAttention_30408368455775
// MI455X (gfx1250) — matrix operand order repaired and checked by reading, not compiled, not run on hardware
//
#include <hip/hip_runtime.h>
#include <hip/hip_bf16.h>

#define B_  4
#define S_  1024
#define D_  1024
#define H_  16
#define DH_ 64

typedef __attribute__((ext_vector_type(16))) __bf16 v16bf;
typedef __attribute__((ext_vector_type(8)))  __bf16 v8bf;
typedef __attribute__((ext_vector_type(4)))  __bf16 v4bf;
typedef __attribute__((ext_vector_type(8)))  float  v8f;

union Frag16 { v16bf v; v8bf h[2]; };

static __device__ inline v8f zero8() {
    v8f z = {0.f,0.f,0.f,0.f,0.f,0.f,0.f,0.f};
    return z;
}

static __device__ inline v8f wmma_bf16(const Frag16& a, const Frag16& b, v8f c) {
    return __builtin_amdgcn_wmma_f32_16x16x32_bf16(
        false, a.v, false, b.v, (short)0, c, false, false);
}

// ---- CDNA5 async copy: global -> LDS, 16B per lane, tracked by ASYNCcnt ----
// LDS flat aperture keeps the wave-relative LDS byte offset in addr[31:0]
// (ISA 10.2 aperture mapping), so truncating the generic pointer is the DS addr.
static __device__ inline void async_to_lds_b128(const __bf16* lds_ptr,
                                                const __bf16* gptr) {
    unsigned loff = (unsigned)(unsigned long long)lds_ptr;
    asm volatile("global_load_async_to_lds_b128 %0, %1, off"
                 :: "v"(loff), "v"(gptr) : "memory");
}
static __device__ inline void wait_async0() {
    asm volatile("s_wait_asynccnt 0" ::: "memory");
}

// reduce across the 16 lanes sharing one matrix row (xor masks 1..8 stay in half)
static __device__ inline float rowmax16(float x) {
    #pragma unroll
    for (int m = 8; m >= 1; m >>= 1) x = fmaxf(x, __shfl_xor(x, m, 32));
    return x;
}
static __device__ inline float rowsum16(float x) {
    #pragma unroll
    for (int m = 8; m >= 1; m >>= 1) x += __shfl_xor(x, m, 32);
    return x;
}

// ---------------------------------------------------------------- converts
__global__ __launch_bounds__(256)
void cvt_f32_bf16(const float* __restrict__ src, __bf16* __restrict__ dst, int n4) {
    int i = blockIdx.x * blockDim.x + threadIdx.x;
    if (i < n4) {
        float4 f = ((const float4*)src)[i];
        v4bf o;
        o[0] = (__bf16)f.x; o[1] = (__bf16)f.y;
        o[2] = (__bf16)f.z; o[3] = (__bf16)f.w;
        ((v4bf*)dst)[i] = o;
    }
}

// W[k][n] f32 -> Wt[n][k] bf16 via LDS-tiled transpose (coalesced both ways)
__global__ __launch_bounds__(256)
void cvt_transpose_w(const float* __restrict__ W, __bf16* __restrict__ Wt) {
    __shared__ float tile[32][33];
    const int tx = threadIdx.x & 31;
    const int ty = threadIdx.x >> 5;          // 0..7
    const int k0 = blockIdx.x * 32;
    const int n0 = blockIdx.y * 32;
    #pragma unroll
    for (int j = 0; j < 4; ++j)
        tile[ty + 8 * j][tx] = W[(size_t)(k0 + ty + 8 * j) * D_ + n0 + tx];
    __syncthreads();
    #pragma unroll
    for (int j = 0; j < 4; ++j)
        Wt[(size_t)(n0 + ty + 8 * j) * D_ + k0 + tx] = (__bf16)tile[tx][ty + 8 * j];
}

// ---------------------------------------------------------------- projections
struct Bias6 { const float* p[6]; };

// C[4096x1024] = Xb @ Wt[p]^T + bias[p].  Block tile 128x128, BK=32, 8 waves,
// each wave 32x64 = 2x4 WMMA accs.  Double-buffered LDS fed by async DMA.
// Q/K projections (p=0,1,3,4) written [B,H,S,DH]; V projections (p=2,5)
// written [B,H,DH,S] so attention V-tiles stage without transpose.
__global__ __launch_bounds__(256)
void proj_gemm(const __bf16* __restrict__ Xb, const __bf16* __restrict__ Wt_all,
               Bias6 biases, __bf16* __restrict__ Heads)
{
    const int p = blockIdx.z;
    const __bf16* Wt   = Wt_all + (size_t)p * D_ * D_;   // [n][k]
    const float*  bias = biases.p[p];
    __bf16* Hd = Heads + (size_t)p * (B_ * S_ * D_);
    const bool vlayout = (p == 2 || p == 5);

    __shared__ __bf16 As[2][128][40];   // [m][k]
    __shared__ __bf16 Bs[2][128][40];   // [n][k]

    const int t       = threadIdx.x;
    const int lane    = t & 31;
    const int wid     = t >> 5;
    const int halfsel = lane >> 4;
    const int ncol    = lane & 15;
    const int waveRow = wid >> 1;
    const int waveCol = wid & 1;
    const int m0 = blockIdx.y * 128;
    const int n0 = blockIdx.x * 128;

    // staging: 128x32 bf16 per matrix = 512 x 16B chunks; 2 chunks each
    const int sr = t >> 1;              // 0..127
    const int sk = (t & 1) * 16;        // 0 or 16

    auto stage = [&](int buf, int k0) {
        #pragma unroll
        for (int c = 0; c < 2; ++c) {
            async_to_lds_b128(&As[buf][sr][sk + c * 8],
                              Xb + (size_t)(m0 + sr) * D_ + k0 + sk + c * 8);
            async_to_lds_b128(&Bs[buf][sr][sk + c * 8],
                              Wt + (size_t)(n0 + sr) * D_ + k0 + sk + c * 8);
        }
    };

    v8f acc[2][4];
    #pragma unroll
    for (int i = 0; i < 2; ++i)
        #pragma unroll
        for (int j = 0; j < 4; ++j) acc[i][j] = zero8();

    stage(0, 0);
    wait_async0();
    __syncthreads();

    int buf = 0;
    for (int k0 = 0; k0 < D_; k0 += 32) {
        if (k0 + 32 < D_) stage(buf ^ 1, k0 + 32);

        Frag16 a[2], b[4];
        #pragma unroll
        for (int mt = 0; mt < 2; ++mt) {
            int r  = waveRow * 32 + mt * 16 + ncol;
            int kb = halfsel * 8;
            a[mt].h[0] = *(const v8bf*)&As[buf][r][kb];
            a[mt].h[1] = *(const v8bf*)&As[buf][r][kb + 16];
        }
        #pragma unroll
        for (int nt = 0; nt < 4; ++nt) {
            int n  = waveCol * 64 + nt * 16 + ncol;
            int kb = halfsel * 8;
            b[nt].h[0] = *(const v8bf*)&Bs[buf][n][kb];
            b[nt].h[1] = *(const v8bf*)&Bs[buf][n][kb + 16];
        }
        #pragma unroll
        for (int mt = 0; mt < 2; ++mt)
            #pragma unroll
            for (int nt = 0; nt < 4; ++nt)
                acc[mt][nt] = wmma_bf16(a[mt], b[nt], acc[mt][nt]);

        if (k0 + 32 < D_) { wait_async0(); __syncthreads(); }
        buf ^= 1;
    }

    // epilogue: bias add, bf16 store in head layout
    #pragma unroll
    for (int mt = 0; mt < 2; ++mt)
        #pragma unroll
        for (int nt = 0; nt < 4; ++nt)
            #pragma unroll
            for (int i = 0; i < 8; ++i) {
                int m = m0 + waveRow * 32 + mt * 16 + i + 8 * halfsel;
                int n = n0 + waveCol * 64 + nt * 16 + ncol;
                float val = acc[mt][nt][i] + bias[n];
                int bb = m >> 10, s = m & 1023;
                int hh = n >> 6,  dh = n & 63;
                size_t idx = vlayout
                    ? ((((size_t)(bb * H_ + hh)) * DH_ + dh) << 10) + s
                    : ((((size_t)(bb * H_ + hh)) << 10) + s) * DH_ + dh;
                Hd[idx] = (__bf16)val;
            }
}

// ---------------------------------------------------------------- attention
// One WG = 128 query rows of one (b,h); per wave a 16-row Q tile.
// Flash attention over 32-key blocks; K/V tiles double-buffered in LDS via
// async DMA (V already transposed to [dh][s] by proj_gemm).
__global__ __launch_bounds__(256)
void attn_fused(const __bf16* __restrict__ Heads, const float* __restrict__ attnw,
                float* __restrict__ out, int variant)
{
    const size_t HT = (size_t)B_ * S_ * D_;
    const __bf16* Qh = Heads + (size_t)(variant * 3 + 0) * HT;  // [B,H,S,DH]
    const __bf16* Kh = Heads + (size_t)(variant * 3 + 1) * HT;  // [B,H,S,DH]
    const __bf16* Vh = Heads + (size_t)(variant * 3 + 2) * HT;  // [B,H,DH,S]

    __shared__ __bf16 Ks[2][32][72];    // [key][dh]
    __shared__ __bf16 Vs[2][64][40];    // [dh][key]
    __shared__ __bf16 Ps[8][16][40];    // per-wave P bounce (C->A layout)

    const int t       = threadIdx.x;
    const int lane    = t & 31;
    const int wid     = t >> 5;
    const int halfsel = lane >> 4;
    const int ncol    = lane & 15;
    const int bh      = blockIdx.y;
    const int q0      = blockIdx.x * 128;
    const int srow    = q0 + wid * 16;

    // staging geometry: K 32x64 = 256 chunks, V 64x32 = 256 chunks, 1+1/thread
    const int kkey = t >> 3, kcg = (t & 7) * 8;   // K: key row, dh chunk
    const int vdh  = t >> 2, vcg = (t & 3) * 8;   // V: dh row, key chunk

    auto stage = [&](int buf, int j0) {
        async_to_lds_b128(&Ks[buf][kkey][kcg],
                          Kh + ((size_t)bh * S_ + j0 + kkey) * DH_ + kcg);
        async_to_lds_b128(&Vs[buf][vdh][vcg],
                          Vh + (((size_t)bh * DH_ + vdh) << 10) + j0 + vcg);
    };

    // Q A-fragments (two K=32 chunks over DH), pre-scaled by 1/sqrt(64)
    Frag16 qa[2];
    {
        int sq = srow + ncol;
        const __bf16* qp = Qh + ((size_t)bh * S_ + sq) * DH_;
        int kb = halfsel * 8;
        #pragma unroll
        for (int c = 0; c < 2; ++c) {
            qa[c].h[0] = *(const v8bf*)(qp + c * 32 + kb);
            qa[c].h[1] = *(const v8bf*)(qp + c * 32 + kb + 16);
            #pragma unroll
            for (int e = 0; e < 16; ++e)
                qa[c].v[e] = (__bf16)((float)qa[c].v[e] * 0.125f);
        }
    }

    v8f O[4];
    #pragma unroll
    for (int n = 0; n < 4; ++n) O[n] = zero8();
    float mrow[8], lrow[8];
    #pragma unroll
    for (int i = 0; i < 8; ++i) { mrow[i] = -1e30f; lrow[i] = 0.0f; }

    stage(0, 0);
    wait_async0();
    __syncthreads();

    int buf = 0;
    for (int j0 = 0; j0 < S_; j0 += 32) {
        if (j0 + 32 < S_) stage(buf ^ 1, j0 + 32);

        // scores: Q(16x64) @ K^T -> two 16x16 key tiles
        v8f s0 = zero8(), s1 = zero8();
        #pragma unroll
        for (int c = 0; c < 2; ++c) {
            Frag16 kb0, kb1;
            int kb = c * 32 + halfsel * 8;
            kb0.h[0] = *(const v8bf*)&Ks[buf][ncol][kb];
            kb0.h[1] = *(const v8bf*)&Ks[buf][ncol][kb + 16];
            kb1.h[0] = *(const v8bf*)&Ks[buf][16 + ncol][kb];
            kb1.h[1] = *(const v8bf*)&Ks[buf][16 + ncol][kb + 16];
            s0 = wmma_bf16(qa[c], kb0, s0);
            s1 = wmma_bf16(qa[c], kb1, s1);
        }

        // online softmax; write exp(P) to per-wave LDS in A-matrix order
        #pragma unroll
        for (int i = 0; i < 8; ++i) {
            float v0 = s0[i], v1 = s1[i];
            float tmax = rowmax16(fmaxf(v0, v1));
            float mnew = fmaxf(mrow[i], tmax);
            float corr = __expf(mrow[i] - mnew);
            float e0 = __expf(v0 - mnew);
            float e1 = __expf(v1 - mnew);
            float rs = rowsum16(e0 + e1);
            lrow[i] = lrow[i] * corr + rs;
            mrow[i] = mnew;
            #pragma unroll
            for (int n = 0; n < 4; ++n) O[n][i] *= corr;
            int M = i + 8 * halfsel;
            Ps[wid][M][ncol]      = (__bf16)e0;
            Ps[wid][M][ncol + 16] = (__bf16)e1;
        }

        // P(16x32) @ V(32x64)
        Frag16 pa;
        {
            int kb = halfsel * 8;
            pa.h[0] = *(const v8bf*)&Ps[wid][ncol][kb];
            pa.h[1] = *(const v8bf*)&Ps[wid][ncol][kb + 16];
        }
        #pragma unroll
        for (int n = 0; n < 4; ++n) {
            Frag16 vb;
            int dh = n * 16 + ncol;
            int kb = halfsel * 8;
            vb.h[0] = *(const v8bf*)&Vs[buf][dh][kb];
            vb.h[1] = *(const v8bf*)&Vs[buf][dh][kb + 16];
            O[n] = wmma_bf16(pa, vb, O[n]);
        }

        if (j0 + 32 < S_) { wait_async0(); __syncthreads(); }
        buf ^= 1;
    }

    // softmax(attn_w) combine weight
    float a0 = attnw[0], a1 = attnw[1];
    float mx = fmaxf(a0, a1);
    float e0 = __expf(a0 - mx), e1 = __expf(a1 - mx);
    float w = (variant == 0 ? e0 : e1) / (e0 + e1);
    int bb = bh >> 4, hh = bh & 15;

    #pragma unroll
    for (int n = 0; n < 4; ++n)
        #pragma unroll
        for (int i = 0; i < 8; ++i) {
            int s  = srow + i + 8 * halfsel;
            int dh = n * 16 + ncol;
            float val = w * (O[n][i] / lrow[i]);
            size_t idx = ((size_t)(bb * S_ + s)) * D_ + hh * DH_ + dh;
            if (variant) out[idx] += val;
            else         out[idx]  = val;
        }
}

// ---------------------------------------------------------------- launch
extern "C" void kernel_launch(void* const* d_in, const int* in_sizes, int n_in,
                              void* d_out, int out_size, void* d_ws, size_t ws_size,
                              hipStream_t stream)
{
    const float* hidden = (const float*)d_in[0];
    const float* W[6];
    Bias6 bs;
    for (int i = 0; i < 6; ++i) {
        W[i]    = (const float*)d_in[1 + 2 * i];
        bs.p[i] = (const float*)d_in[2 + 2 * i];
    }
    const float* attnw = (const float*)d_in[13];
    float* out = (float*)d_out;

    // workspace (bf16): Xb 8MB | Wt 12MB | Heads 48MB  (= 68MB total)
    char* ws = (char*)d_ws;
    __bf16* Xb    = (__bf16*)ws;
    __bf16* Wt    = (__bf16*)(ws + (size_t)8  * 1024 * 1024);
    __bf16* Heads = (__bf16*)(ws + (size_t)20 * 1024 * 1024);

    const int NX4 = (B_ * S_ * D_) / 4;
    cvt_f32_bf16<<<(NX4 + 255) / 256, 256, 0, stream>>>(hidden, Xb, NX4);
    for (int i = 0; i < 6; ++i)
        cvt_transpose_w<<<dim3(32, 32), 256, 0, stream>>>(
            W[i], Wt + (size_t)i * D_ * D_);

    proj_gemm<<<dim3(D_ / 128, (B_ * S_) / 128, 6), 256, 0, stream>>>(Xb, Wt, bs, Heads);

    attn_fused<<<dim3(S_ / 128, B_ * H_), 256, 0, stream>>>(Heads, attnw, out, 0);
    attn_fused<<<dim3(S_ / 128, B_ * H_), 256, 0, stream>>>(Heads, attnw, out, 1);
}
